// RelationClassifier_ent1_ent2_average_pooled_20272245637486
// MI455X (gfx1250) — matrix operand order repaired and checked by reading, not compiled, not run on hardware
//
#include <hip/hip_runtime.h>
#include <hip/hip_bf16.h>

// Problem constants (from reference): B=64, S=2048, H=768
#define BATCH 64
#define SEQ   2048
#define HID   768
#define K2H   1536   // 2*H

#define ENT1_START 1
#define ENT1_END   2
#define ENT2_START 3
#define ENT2_END   4

typedef __attribute__((ext_vector_type(16))) _Float16 v16h;
typedef __attribute__((ext_vector_type(8)))  float    v8f;

// ---------------------------------------------------------------------------
// Kernel 1: one workgroup (8 wave32) per batch.
//  - find first occurrence of the 4 marker ids via shared atomicMin
//  - mean-pool sequence_output rows strictly inside (s,e) for both spans
//  - write combined[b, 0:768]=ent1, combined[b, 768:1536]=ent2 as FP16
// ---------------------------------------------------------------------------
__global__ __launch_bounds__(256) void span_pool_kernel(
    const int*   __restrict__ ids,       // [B,S]
    const float* __restrict__ reps,      // [B,S,H] f32
    _Float16*    __restrict__ combined)  // [B,2H] f16 (workspace)
{
    __shared__ int sh[4];                // s1, e1, s2, e2 (first occurrence)
    const int b   = blockIdx.x;
    const int tid = threadIdx.x;
    const int SENT = 0x7fffffff;

    if (tid < 4) sh[tid] = SENT;
    __syncthreads();

    const int* idrow = ids + (size_t)b * SEQ;
    for (int p = tid; p < SEQ; p += 256) {
        int t = idrow[p];
        if      (t == ENT1_START) atomicMin(&sh[0], p);
        else if (t == ENT1_END)   atomicMin(&sh[1], p);
        else if (t == ENT2_START) atomicMin(&sh[2], p);
        else if (t == ENT2_END)   atomicMin(&sh[3], p);
    }
    __syncthreads();

    const int s1 = sh[0], e1 = sh[1], s2 = sh[2], e2 = sh[3];
    // cnt = #positions with s < p < e ; valid only if both markers found
    const int cnt1 = (s1 != SENT && e1 != SENT && e1 > s1 + 1) ? (e1 - s1 - 1) : 0;
    const int cnt2 = (s2 != SENT && e2 != SENT && e2 > s2 + 1) ? (e2 - s2 - 1) : 0;
    const float inv1 = (cnt1 > 0) ? 1.0f / (float)cnt1 : 0.0f;
    const float inv2 = (cnt2 > 0) ? 1.0f / (float)cnt2 : 0.0f;

    const float* base = reps + (size_t)b * SEQ * HID;
    _Float16* crow = combined + (size_t)b * K2H;

    // H=768, 256 threads -> 3 channels each; span loops are ~1-15 iters.
    for (int h = tid; h < HID; h += 256) {
        float sum1 = 0.0f;
        if (cnt1 > 0) {
            for (int p = s1 + 1; p < e1; ++p)
                sum1 += base[(size_t)p * HID + h];   // coalesced across h
        }
        float sum2 = 0.0f;
        if (cnt2 > 0) {
            for (int p = s2 + 1; p < e2; ++p)
                sum2 += base[(size_t)p * HID + h];
        }
        crow[h]       = (_Float16)(sum1 * inv1);
        crow[HID + h] = (_Float16)(sum2 * inv2);
    }
}

// ---------------------------------------------------------------------------
// Kernel 2: WMMA GEMV  out[64] = combined(64x1536) . W(1536) + bias
// One block, 4 wave32; wave w owns batch rows [16w, 16w+16).
// K swept in 48 chunks of 32 with v_wmma_f32_16x16x32_f16; W is broadcast
// into every N-column of the B fragment, so D[:,0] is the dot product.
// ---------------------------------------------------------------------------
__global__ __launch_bounds__(128) void gemv_wmma_kernel(
    const _Float16* __restrict__ combined,  // [B,2H] f16
    const float*    __restrict__ W,         // [2H] f32
    const float*    __restrict__ bias,      // [1]  f32
    float*          __restrict__ out)       // [B]  f32
{
    const int lane = threadIdx.x & 31;
    const int wave = threadIdx.x >> 5;
    const int g    = lane >> 4;    // lane-group (0: lanes 0-15, 1: lanes 16-31)
    const int m    = lane & 15;    // row within 16x16 tile
    const int row  = wave * 16 + m;

    const _Float16* arow = combined + (size_t)row * K2H;

    v8f acc = {};
    for (int kc = 0; kc < K2H; kc += 32) {
        // A fragment: 16x32 f16, lane holds row `m`; per ISA 7.12.2 the
        // half-pair in VGPR v covers K = (v<4 ? 2v : 16+2(v-4)) + 8g.
        v16h a;
        #pragma unroll
        for (int v = 0; v < 8; ++v) {
            const int pk = ((v < 4) ? (2 * v) : (16 + 2 * (v - 4))) + 8 * g;
            a[2 * v]     = arow[kc + pk];
            a[2 * v + 1] = arow[kc + pk + 1];
        }
        // B fragment: 32x16 f16, half i of this lane is K = (i<8 ? i : 16+(i-8)) + 8g, N = m.
        // Broadcast W[k] across all N columns (value independent of m).
        v16h bfrag;
        const float* wp = W + kc + 8 * g;
        #pragma unroll
        for (int i = 0; i < 16; ++i)
            bfrag[i] = (_Float16)wp[(i < 8) ? i : (i + 8)];

        acc = __builtin_amdgcn_wmma_f32_16x16x32_f16(
            /*neg_a=*/false, a, /*neg_b=*/false, bfrag,
            /*c_mod=*/(short)0, acc, /*reuse_a=*/false, /*reuse_b=*/false);
    }

    // D layout: N = lane%16, M = r + 8*(lane/16). Column N==0 lives in
    // lanes 0 (M=0..7) and 16 (M=8..15) of each wave.
    if (m == 0) {
        const float bb = bias[0];
        #pragma unroll
        for (int r = 0; r < 8; ++r)
            out[wave * 16 + g * 8 + r] = acc[r] + bb;
    }
}

extern "C" void kernel_launch(void* const* d_in, const int* in_sizes, int n_in,
                              void* d_out, int out_size, void* d_ws, size_t ws_size,
                              hipStream_t stream) {
    const int*   ids  = (const int*)  d_in[0];   // input_ids [B,S]
    // d_in[1] = attention_mask (unused by the reference computation)
    const float* reps = (const float*)d_in[2];   // sequence_output [B,S,H]
    const float* W    = (const float*)d_in[3];   // [1,2H]
    const float* bias = (const float*)d_in[4];   // [1]
    float*       out  = (float*)d_out;           // [B]

    _Float16* combined = (_Float16*)d_ws;        // needs B*2H*2 = 192 KiB

    span_pool_kernel<<<BATCH, 256, 0, stream>>>(ids, reps, combined);
    gemv_wmma_kernel<<<1, 128, 0, stream>>>(combined, W, bias, out);
}
